// GroupedQueryAttention_39633958207636
// MI455X (gfx1250) — matrix operand order repaired and checked by reading, not compiled, not run on hardware
//
#include <hip/hip_runtime.h>

// ---------------------------------------------------------------------------
// GQA (grouped-query attention) for MI455X / gfx1250, bf16 WMMA pipeline.
// WMMA fragments = pairs of 16B-aligned LDS vector loads; global->LDS staging
// uses gfx1250 async-to-LDS (ASYNCcnt) when the builtins are available.
// ---------------------------------------------------------------------------

typedef __bf16 bf16_t;
typedef __attribute__((ext_vector_type(4)))  __bf16 v4bf;
typedef __attribute__((ext_vector_type(8)))  __bf16 v8bf;
typedef __attribute__((ext_vector_type(16))) __bf16 v16bf;
typedef __attribute__((ext_vector_type(8)))  float  v8f;
typedef __attribute__((ext_vector_type(4)))  int    vi4;

union Frag16 { v16bf v; uint4 q[2]; };   // 32B WMMA operand
union Vec8bf { v8bf  v; uint4 q;    };   // 16B staging vector

constexpr int D_MODEL = 2048;
constexpr int H       = 32;
constexpr int G       = 8;
constexpr int DK      = 64;
constexpr int BATCH   = 2;
constexpr int SEQ     = 2048;
constexpr int MTOT    = BATCH * SEQ;     // 4096
constexpr int KVDIM   = G * DK;          // 512

// ---- gfx1250 async global->LDS copy (ASYNCcnt), with sync fallback --------
#if defined(__HIP_DEVICE_COMPILE__)
#if __has_builtin(__builtin_amdgcn_global_load_async_to_lds_b128) && \
    __has_builtin(__builtin_amdgcn_s_wait_asynccnt)
#define GQA_ASYNC 1
#endif
#endif
#ifndef GQA_ASYNC
#define GQA_ASYNC 0
#endif

__device__ __forceinline__ void async_copy16(const bf16_t* g, bf16_t* l) {
#if GQA_ASYNC
    __builtin_amdgcn_global_load_async_to_lds_b128(
        (__attribute__((address_space(1))) vi4*)g,
        (__attribute__((address_space(3))) vi4*)l, 0, 0);
#else
    *reinterpret_cast<uint4*>(l) = *reinterpret_cast<const uint4*>(g);
#endif
}
__device__ __forceinline__ void async_wait_all() {
#if GQA_ASYNC
    __builtin_amdgcn_s_wait_asynccnt(0);
#endif
}

// ---------------------------------------------------------------------------
// fp32 -> bf16 cast (for activations x)
// ---------------------------------------------------------------------------
__global__ __launch_bounds__(256)
void cast_f32_bf16(const float* __restrict__ src, bf16_t* __restrict__ dst, int n4) {
    int i = blockIdx.x * blockDim.x + threadIdx.x;
    if (i < n4) {
        float4 v = reinterpret_cast<const float4*>(src)[i];
        v4bf o;
        o[0] = (bf16_t)v.x; o[1] = (bf16_t)v.y;
        o[2] = (bf16_t)v.z; o[3] = (bf16_t)v.w;
        *reinterpret_cast<v4bf*>(dst + 4 * i) = o;
    }
}

// ---------------------------------------------------------------------------
// fp32 -> bf16 cast + transpose: W[K][N] -> Wt[N][K], 64x64 tiles via LDS
// ---------------------------------------------------------------------------
__global__ __launch_bounds__(256)
void transpose_cast_f32_bf16(const float* __restrict__ W, bf16_t* __restrict__ Wt,
                             int K, int N) {
    __shared__ bf16_t T[64 * 72];        // [n][k]
    const int tid = threadIdx.x;
    const int k0  = blockIdx.y * 64;
    const int n0  = blockIdx.x * 64;
    #pragma unroll
    for (int v = 0; v < 4; ++v) {
        int vid = tid + v * 256;          // 1024 float4 = 64x64 f32
        int r   = vid >> 4;               // k row in tile
        int c   = (vid & 15) << 2;        // n col base
        float4 w = *reinterpret_cast<const float4*>(&W[(size_t)(k0 + r) * N + n0 + c]);
        T[(c + 0) * 72 + r] = (bf16_t)w.x;
        T[(c + 1) * 72 + r] = (bf16_t)w.y;
        T[(c + 2) * 72 + r] = (bf16_t)w.z;
        T[(c + 3) * 72 + r] = (bf16_t)w.w;
    }
    __syncthreads();
    #pragma unroll
    for (int v = 0; v < 2; ++v) {
        int vid = tid + v * 256;          // 512 vec8
        int rn  = vid >> 3;
        int ck  = (vid & 7) << 3;
        *reinterpret_cast<uint4*>(&Wt[(size_t)(n0 + rn) * K + k0 + ck]) =
            *reinterpret_cast<const uint4*>(&T[rn * 72 + ck]);
    }
}

// ---------------------------------------------------------------------------
// bf16 GEMM, f32 accum: C[M,N] = A[M,K] @ Bt[N,K]^T + bias[N]
// Block tile 128x128, K-step 32, 8 waves (4x2), wave tile 32x64 (2x4 WMMA).
// Double-buffered LDS; next tile staged via async-to-LDS while computing.
// ---------------------------------------------------------------------------
template <bool OUT_BF16>
__global__ __launch_bounds__(256)
void gemm_bf16_wmma(const bf16_t* __restrict__ A, const bf16_t* __restrict__ Bt,
                    const float* __restrict__ bias, void* __restrict__ Cp,
                    int M, int N, int K) {
    constexpr int BM = 128, BN = 128, BK = 32;
    constexpr int LD = BK + 8;            // 40 bf16 (80B rows: 16B aligned)
    __shared__ bf16_t As[2][BM * LD];
    __shared__ bf16_t Bs[2][BN * LD];

    const int tid  = threadIdx.x;
    const int wave = tid >> 5;
    const int lane = tid & 31;
    const int wm   = wave >> 1;           // 0..3 -> 32-row strip
    const int wn   = wave & 1;            // 0..1 -> 64-col strip
    const int l16  = lane & 15;
    const int hi   = lane >> 4;

    const int blockM = blockIdx.y * BM;
    const int blockN = blockIdx.x * BN;

    v8f acc[2][4] = {{{}, {}, {}, {}}, {{}, {}, {}, {}}};

    auto stage = [&](int buf, int kt) {
        #pragma unroll
        for (int v = 0; v < 2; ++v) {
            int vid = tid + v * 256;
            int r   = vid >> 2;
            int c   = (vid & 3) << 3;
            async_copy16(&A[(size_t)(blockM + r) * K + kt + c], &As[buf][r * LD + c]);
            async_copy16(&Bt[(size_t)(blockN + r) * K + kt + c], &Bs[buf][r * LD + c]);
        }
    };

    stage(0, 0);
    async_wait_all();
    __syncthreads();

    const int nk = K / BK;
    for (int ik = 0; ik < nk; ++ik) {
        const int buf = ik & 1;
        if (ik + 1 < nk)                   // overlap next-tile copy with compute
            stage(buf ^ 1, (ik + 1) * BK);

        // A fragment (ISA 7.12.2 16-bit A 16x32): two contiguous 16B runs/lane
        Frag16 af[2];
        #pragma unroll
        for (int i = 0; i < 2; ++i) {
            const int m0 = (wm * 32 + i * 16 + l16) * LD;
            af[i].q[0] = *reinterpret_cast<const uint4*>(&As[buf][m0 + (hi << 3)]);
            af[i].q[1] = *reinterpret_cast<const uint4*>(&As[buf][m0 + 16 + (hi << 3)]);
        }
        // B fragment: same K order as A (two 16B runs at hi*8 and 16+hi*8)
        Frag16 bfr[4];
        #pragma unroll
        for (int j = 0; j < 4; ++j) {
            const int nb = (wn * 64 + j * 16 + l16) * LD + (hi << 3);
            bfr[j].q[0] = *reinterpret_cast<const uint4*>(&Bs[buf][nb]);
            bfr[j].q[1] = *reinterpret_cast<const uint4*>(&Bs[buf][nb + 16]);
        }
        #pragma unroll
        for (int i = 0; i < 2; ++i)
            #pragma unroll
            for (int j = 0; j < 4; ++j)
                acc[i][j] = __builtin_amdgcn_wmma_f32_16x16x32_bf16(
                    false, af[i].v, false, bfr[j].v, (short)0, acc[i][j], false, false);

        async_wait_all();                  // own async copies complete...
        __syncthreads();                   // ...then join waves -> buffer ready
    }

    // C/D layout: VGPR r -> row r (lanes 0-15) / r+8 (lanes 16-31); col = lane&15
    #pragma unroll
    for (int i = 0; i < 2; ++i)
        #pragma unroll
        for (int j = 0; j < 4; ++j) {
            const int col = blockN + wn * 64 + j * 16 + l16;
            const float bv = bias[col];
            #pragma unroll
            for (int r = 0; r < 8; ++r) {
                const int row = blockM + wm * 32 + i * 16 + r + (hi << 3);
                const float val = acc[i][j][r] + bv;
                if (OUT_BF16)
                    reinterpret_cast<bf16_t*>(Cp)[(size_t)row * N + col] = (bf16_t)val;
                else
                    reinterpret_cast<float*>(Cp)[(size_t)row * N + col] = val;
            }
        }
}

// ---------------------------------------------------------------------------
// Flash-style GQA attention. Grid: (SEQ/64, BATCH*H). Block: 128 threads.
// Each wave owns a 16-query strip; block streams 64-key K/V slabs via LDS.
// K staged with async-to-LDS; V staged transposed (Vt[d][key]) so PV
// B-fragments are contiguous.
// ---------------------------------------------------------------------------
__global__ __launch_bounds__(128)
void gqa_attention_wmma(const bf16_t* __restrict__ Qb, const bf16_t* __restrict__ Kb,
                        const bf16_t* __restrict__ Vb, bf16_t* __restrict__ Ob) {
    constexpr int LDT = 72;               // 64 + 8 pad (144B rows)
    __shared__ bf16_t Qs[64 * LDT];       // [query][d]
    __shared__ bf16_t Ks[64 * LDT];       // [key][d]
    __shared__ bf16_t Vt[64 * LDT];       // [d][key]   (transposed)
    __shared__ bf16_t Ps[4 * 16 * LDT];   // per-wave P [query][key]

    const int tid  = threadIdx.x;
    const int wave = tid >> 5;
    const int lane = tid & 31;
    const int l16  = lane & 15;
    const int hi   = lane >> 4;

    const int bh = blockIdx.y;            // b*H + h
    const int b  = bh >> 5;
    const int h  = bh & 31;
    const int g  = h >> 2;                // GROUP_SIZE = 4
    const int q0 = blockIdx.x * 64;

    const bf16_t* Qp = Qb + (size_t)b * SEQ * D_MODEL + h * DK;
    const bf16_t* Kp = Kb + (size_t)b * SEQ * KVDIM + g * DK;
    const bf16_t* Vp = Vb + (size_t)b * SEQ * KVDIM + g * DK;

    // stage Q block 64x64 (async)
    #pragma unroll
    for (int v = 0; v < 4; ++v) {
        int vid = tid + v * 128;
        int r   = vid >> 3;
        int c   = (vid & 7) << 3;
        async_copy16(&Qp[(size_t)(q0 + r) * D_MODEL + c], &Qs[r * LDT + c]);
    }
    async_wait_all();
    __syncthreads();

    // Q A-fragments (two K-steps over dk=64), contiguous per lane
    Frag16 qa[2];
    #pragma unroll
    for (int ks = 0; ks < 2; ++ks) {
        const int base = (wave * 16 + l16) * LDT + ks * 32 + (hi << 3);
        qa[ks].q[0] = *reinterpret_cast<const uint4*>(&Qs[base]);
        qa[ks].q[1] = *reinterpret_cast<const uint4*>(&Qs[base + 16]);
    }

    v8f acc[4] = {{}, {}, {}, {}};
    float m_i[8], l_i[8];
    #pragma unroll
    for (int r = 0; r < 8; ++r) { m_i[r] = -1e30f; l_i[r] = 0.0f; }

    const float sm_scale = 0.125f;        // 1/sqrt(64)
    bf16_t* Pw = &Ps[wave * 16 * LDT];

    for (int kt = 0; kt < SEQ; kt += 64) {
        __syncthreads();
        // stage K (async, row-major) and V (transposed via VGPRs) 64x64 slabs
        #pragma unroll
        for (int v = 0; v < 4; ++v) {
            int vid = tid + v * 128;
            int r   = vid >> 3;           // key
            int c   = (vid & 7) << 3;     // d base
            async_copy16(&Kp[(size_t)(kt + r) * KVDIM + c], &Ks[r * LDT + c]);
            Vec8bf u;
            u.q = *reinterpret_cast<const uint4*>(&Vp[(size_t)(kt + r) * KVDIM + c]);
            #pragma unroll
            for (int i = 0; i < 8; ++i)
                Vt[(c + i) * LDT + r] = u.v[i];
        }
        if (kt + 64 < SEQ) {              // L2 prefetch of next slab
            int r = tid >> 3, c = (tid & 7) << 3;
            __builtin_prefetch(&Kp[(size_t)(kt + 64 + r) * KVDIM + c], 0, 0);
            __builtin_prefetch(&Vp[(size_t)(kt + 64 + r) * KVDIM + c], 0, 0);
        }
        async_wait_all();
        __syncthreads();

        // scores = Q @ K^T : four 16-key subtiles x two dk K-steps
        v8f sc[4] = {{}, {}, {}, {}};
        #pragma unroll
        for (int j = 0; j < 4; ++j) {
            #pragma unroll
            for (int ks = 0; ks < 2; ++ks) {
                Frag16 kb;                 // b[e] = K[key = j*16+l16][dk = ks*32 + table order]
                const int base = (j * 16 + l16) * LDT + ks * 32 + (hi << 3);
                kb.q[0] = *reinterpret_cast<const uint4*>(&Ks[base]);
                kb.q[1] = *reinterpret_cast<const uint4*>(&Ks[base + 16]);
                sc[j] = __builtin_amdgcn_wmma_f32_16x16x32_bf16(
                    false, qa[ks].v, false, kb.v, (short)0, sc[j], false, false);
            }
        }

        // online softmax over the 64-key slab
        float cs[8];
        #pragma unroll
        for (int r = 0; r < 8; ++r) {
            float s0 = sc[0][r] * sm_scale, s1 = sc[1][r] * sm_scale;
            float s2 = sc[2][r] * sm_scale, s3 = sc[3][r] * sm_scale;
            float mx = fmaxf(fmaxf(s0, s1), fmaxf(s2, s3));
            #pragma unroll
            for (int off = 1; off < 16; off <<= 1)
                mx = fmaxf(mx, __shfl_xor(mx, off, 16));
            float mn = fmaxf(m_i[r], mx);
            float e0 = __expf(s0 - mn), e1 = __expf(s1 - mn);
            float e2 = __expf(s2 - mn), e3 = __expf(s3 - mn);
            float rs = (e0 + e1) + (e2 + e3);
            #pragma unroll
            for (int off = 1; off < 16; off <<= 1)
                rs += __shfl_xor(rs, off, 16);
            float c = __expf(m_i[r] - mn);
            l_i[r]  = l_i[r] * c + rs;
            m_i[r]  = mn;
            cs[r]   = c;
            const int prow = (r + (hi << 3)) * LDT;
            Pw[prow + l16]      = (bf16_t)e0;
            Pw[prow + 16 + l16] = (bf16_t)e1;
            Pw[prow + 32 + l16] = (bf16_t)e2;
            Pw[prow + 48 + l16] = (bf16_t)e3;
        }

        // reload P as A-fragments (contiguous per lane)
        Frag16 pa[2];
        #pragma unroll
        for (int ks = 0; ks < 2; ++ks) {
            const int base = l16 * LDT + ks * 32 + (hi << 3);
            pa[ks].q[0] = *reinterpret_cast<const uint4*>(&Pw[base]);
            pa[ks].q[1] = *reinterpret_cast<const uint4*>(&Pw[base + 16]);
        }

        // acc = acc*rescale + P @ V (4 dk tiles x 2 key K-steps)
        #pragma unroll
        for (int j = 0; j < 4; ++j) {
            v8f a = acc[j];
            #pragma unroll
            for (int r = 0; r < 8; ++r) a[r] *= cs[r];
            #pragma unroll
            for (int ks = 0; ks < 2; ++ks) {
                Frag16 vb;                 // b[e] = V[key = ks*32 + table order][d = j*16+l16]
                const int base = (j * 16 + l16) * LDT + ks * 32 + (hi << 3);
                vb.q[0] = *reinterpret_cast<const uint4*>(&Vt[base]);
                vb.q[1] = *reinterpret_cast<const uint4*>(&Vt[base + 16]);
                a = __builtin_amdgcn_wmma_f32_16x16x32_bf16(
                    false, pa[ks].v, false, vb.v, (short)0, a, false, false);
            }
            acc[j] = a;
        }
    }

    // normalize and store to att buffer [b, s, h*64 + d] (bf16)
    #pragma unroll
    for (int j = 0; j < 4; ++j)
        #pragma unroll
        for (int r = 0; r < 8; ++r) {
            const int row = q0 + wave * 16 + r + (hi << 3);
            const int col = h * DK + j * 16 + l16;
            Ob[(size_t)(b * SEQ + row) * D_MODEL + col] = (bf16_t)(acc[j][r] / l_i[r]);
        }
}

// ---------------------------------------------------------------------------
// Launcher
// ---------------------------------------------------------------------------
extern "C" void kernel_launch(void* const* d_in, const int* in_sizes, int n_in,
                              void* d_out, int out_size, void* d_ws, size_t ws_size,
                              hipStream_t stream) {
    const float* x  = (const float*)d_in[0];
    const float* Wq = (const float*)d_in[1];
    const float* bq = (const float*)d_in[2];
    const float* Wk = (const float*)d_in[3];
    const float* bk = (const float*)d_in[4];
    const float* Wv = (const float*)d_in[5];
    const float* bv = (const float*)d_in[6];
    const float* Wo = (const float*)d_in[7];
    const float* bo = (const float*)d_in[8];

    // workspace layout (bf16 buffers; weights stored transposed [N][K])
    char* ws = (char*)d_ws;
    bf16_t* xb  = (bf16_t*)ws;  ws += (size_t)MTOT * D_MODEL * 2;
    bf16_t* Wqt = (bf16_t*)ws;  ws += (size_t)D_MODEL * D_MODEL * 2;
    bf16_t* Wkt = (bf16_t*)ws;  ws += (size_t)D_MODEL * KVDIM * 2;
    bf16_t* Wvt = (bf16_t*)ws;  ws += (size_t)D_MODEL * KVDIM * 2;
    bf16_t* Wot = (bf16_t*)ws;  ws += (size_t)D_MODEL * D_MODEL * 2;
    bf16_t* Qb  = (bf16_t*)ws;  ws += (size_t)MTOT * D_MODEL * 2;
    bf16_t* Kb  = (bf16_t*)ws;  ws += (size_t)MTOT * KVDIM * 2;
    bf16_t* Vb  = (bf16_t*)ws;  ws += (size_t)MTOT * KVDIM * 2;
    bf16_t* Ab  = (bf16_t*)ws;  ws += (size_t)MTOT * D_MODEL * 2;

    {   // x: plain cast
        size_t n = (size_t)MTOT * D_MODEL;
        int n4 = (int)(n / 4);
        cast_f32_bf16<<<(n4 + 255) / 256, 256, 0, stream>>>(x, xb, n4);
    }
    // weights: cast + transpose (W[K][N] -> Wt[N][K])
    transpose_cast_f32_bf16<<<dim3(D_MODEL / 64, D_MODEL / 64), 256, 0, stream>>>(
        Wq, Wqt, D_MODEL, D_MODEL);
    transpose_cast_f32_bf16<<<dim3(KVDIM / 64, D_MODEL / 64), 256, 0, stream>>>(
        Wk, Wkt, D_MODEL, KVDIM);
    transpose_cast_f32_bf16<<<dim3(KVDIM / 64, D_MODEL / 64), 256, 0, stream>>>(
        Wv, Wvt, D_MODEL, KVDIM);
    transpose_cast_f32_bf16<<<dim3(D_MODEL / 64, D_MODEL / 64), 256, 0, stream>>>(
        Wo, Wot, D_MODEL, D_MODEL);

    // projections (bf16 out)
    gemm_bf16_wmma<true><<<dim3(D_MODEL / 128, MTOT / 128), 256, 0, stream>>>(
        xb, Wqt, bq, (void*)Qb, MTOT, D_MODEL, D_MODEL);
    gemm_bf16_wmma<true><<<dim3(KVDIM / 128, MTOT / 128), 256, 0, stream>>>(
        xb, Wkt, bk, (void*)Kb, MTOT, KVDIM, D_MODEL);
    gemm_bf16_wmma<true><<<dim3(KVDIM / 128, MTOT / 128), 256, 0, stream>>>(
        xb, Wvt, bv, (void*)Vb, MTOT, KVDIM, D_MODEL);

    // attention: grid (query blocks of 64, B*H heads)
    gqa_attention_wmma<<<dim3(SEQ / 64, BATCH * H), 128, 0, stream>>>(Qb, Kb, Vb, Ab);

    // output projection (f32 out to d_out)
    gemm_bf16_wmma<false><<<dim3(D_MODEL / 128, MTOT / 128), 256, 0, stream>>>(
        Ab, Wot, bo, d_out, MTOT, D_MODEL, D_MODEL);
}
